// SimpleAttention_11836929867810
// MI455X (gfx1250) — compile-verified, matrix operand order checked by reading, not run on hardware
//
#include <hip/hip_runtime.h>
#include <hip/hip_bf16.h>
#include <math.h>

// ---------------------------------------------------------------------------
// SimpleAttention for MI455X (gfx1250): bf16 WMMA + TDM async tensor loads.
//   B=16, S=32, H=8, KVH=1, D=256, HIDDEN=1024, P=4096, T=4128
// Pipeline:
//   1) gemm_bf16 : hs @ {Wq,Wk,Wv}       (fp32 in, bf16 wmma, fp32 out)
//   2) rope      : RoPE(q)->Qw bf16, RoPE(k)->Kn fp32
//   3) attn      : split-KV flash attention, 128 WGs (16 b x 8 kv-chunks),
//                  16 waves/WG; K tile DMA'd to LDS by the Tensor Data Mover
//                  (padded rows for conflict-free ds_load_b128), V staged
//                  transposed as bf16; online softmax; partial (O,m,l) to ws
//   4) combine   : merge 8 chunk partials -> attn output (B,S,H*D) fp32
//   5) gemm_bf16 : attn @ Wo -> d_out
// ---------------------------------------------------------------------------

typedef __attribute__((ext_vector_type(16))) __bf16 v16bf;
typedef __attribute__((ext_vector_type(8)))  __bf16 v8bf;
typedef __attribute__((ext_vector_type(8)))  float  v8f;
typedef __attribute__((ext_vector_type(4)))  unsigned int u32x4;
typedef __attribute__((ext_vector_type(8)))  int i32x8;
typedef __attribute__((ext_vector_type(4)))  int i32x4;

__device__ __forceinline__ __bf16 tob(float x) { return (__bf16)x; }

union frag_u { v16bf v; v8bf h[2]; };

// bf16 fragment per CDNA5 16-bit A/B layout: elems 0..7 = p[0..7], 8..15 = p[16..23]
__device__ __forceinline__ v16bf load_frag(const __bf16* p) {
    frag_u u;
    u.h[0] = *(const v8bf*)(p);
    u.h[1] = *(const v8bf*)(p + 16);
    return u.v;
}

// Same fragment sourced from fp32 LDS (TDM-loaded), converting to bf16 at use.
__device__ __forceinline__ v16bf load_frag_f32(const float* p) {
    v16bf r;
    #pragma unroll
    for (int i = 0; i < 8; i++) { r[i] = tob(p[i]); r[8 + i] = tob(p[16 + i]); }
    return r;
}

__device__ __forceinline__ v8f wmma_bf16(v16bf a, v16bf b, v8f c) {
    return __builtin_amdgcn_wmma_f32_16x16x32_bf16(
        /*neg_a=*/false, a, /*neg_b=*/false, b,
        /*c_mod=*/(short)0, c, /*reuse_a=*/false, /*reuse_b=*/false);
}

// ---------------------------------------------------------------------------
// Tensor Data Mover: DMA a rows x 256 fp32 tile (row stride 256 elems) from
// global memory into LDS, inserting a 4-DWORD pad after every 256 DWORDs so
// the LDS tile has a 260-DWORD row stride (260 % 64 = 4 -> conflict-free
// b128 fragment reads across 16 lanes).
// D# per CDNA5 ISA.
// This toolchain exposes the 6-arg builtin:
//   (uint32x4 g0, int32x8 g1, int32x4 g2, int32x4 g3, int32x8 extra, int cpol)
// ---------------------------------------------------------------------------
#define HAVE_TDM (__has_builtin(__builtin_amdgcn_tensor_load_to_lds))

__device__ __forceinline__ void tdm_load_k_tile(unsigned int lds_byte_addr,
                                                const float* gsrc,
                                                unsigned int rows) {
#if HAVE_TDM
    unsigned long long ga = (unsigned long long)(const void*)gsrc;
    u32x4 g0;
    g0[0] = 1u;                                            // count=1, user mode
    g0[1] = lds_byte_addr;                                 // lds_addr
    g0[2] = (unsigned int)(ga & 0xffffffffull);            // global_addr[31:0]
    g0[3] = (unsigned int)((ga >> 32) & 0x01ffffffull)     // global_addr[56:32]
          | (2u << 30);                                    // type = 2 ("image")
    i32x8 g1;
    g1[0] = (int)((2u << 16)      // data_size = 4 bytes
                | (1u << 20)      // pad_enable
                | (7u << 22)      // pad_interval: every 256 DWORDs
                | (3u << 25));    // pad_amount: 4 DWORDs
    g1[1] = (int)(256u << 16);    // tensor_dim0 = 256 (bits 79:48, low part)
    g1[2] = (int)(rows << 16);    // tensor_dim1 = rows (bits 95:80)
    g1[3] = (int)(256u << 16);    // tile_dim0 = 256 (bits 127:112)
    g1[4] = (int)rows;            // tile_dim1 = rows (bits 143:128); tile_dim2=0
    g1[5] = 256;                  // tensor_dim0_stride = 256 (bits 191:160)
    g1[6] = 0;
    g1[7] = 0;
    i32x4 gz4 = {};
    i32x8 gz8 = {};
    __builtin_amdgcn_tensor_load_to_lds(g0, g1, gz4, gz4, gz8, 0);
#else
    (void)lds_byte_addr; (void)gsrc; (void)rows;
#endif
}

__device__ __forceinline__ unsigned int lds_addr_of(const void* p) {
    // generic pointer to LDS: low 32 bits are the LDS byte address
    return (unsigned int)(unsigned long long)p;
}

// ---------------------------------------------------------------------------
// Generic tiled GEMM: C[M,N] = A[M,K] @ B[K,N], fp32 in/out, bf16 compute.
// Workgroup tile 128(M) x 64(N); 8 waves, each wave = 64x16 (4 C tiles).
// Requires M%128==0, N%64==0, K%32==0 (true for all calls here).
// ---------------------------------------------------------------------------
__global__ __launch_bounds__(256) void gemm_bf16_kernel(
    const float* __restrict__ A, const float* __restrict__ B,
    float* __restrict__ C, int M, int N, int K)
{
    __shared__ __bf16 Al[128][40];
    __shared__ __bf16 Bl[64][40];   // stored transposed: Bl[n][k]

    const int tid = threadIdx.x;
    const int w = tid >> 5, l = tid & 31;
    const int lane16 = l & 15, lhalf = l >> 4;
    const int m0 = blockIdx.x * 128, n0 = blockIdx.y * 64;
    const int wm = (w & 1) * 64, wn = (w >> 1) * 16;

    v8f acc[4] = {};

    for (int k0 = 0; k0 < K; k0 += 32) {
        __syncthreads();
        // ---- stage A tile (128 x 32), convert fp32 -> bf16
        {
            const int row = tid >> 1, ks = (tid & 1) * 16;
            const float* src = A + (size_t)(m0 + row) * K + k0 + ks;
            float tf[16];
            #pragma unroll
            for (int i = 0; i < 4; i++) {
                float4 f = ((const float4*)src)[i];
                tf[4*i+0] = f.x; tf[4*i+1] = f.y; tf[4*i+2] = f.z; tf[4*i+3] = f.w;
            }
            v8bf a0, a1;
            #pragma unroll
            for (int i = 0; i < 8; i++) { a0[i] = tob(tf[i]); a1[i] = tob(tf[8+i]); }
            *(v8bf*)&Al[row][ks]     = a0;
            *(v8bf*)&Al[row][ks + 8] = a1;
        }
        // ---- stage B tile (32 x 64) transposed, convert fp32 -> bf16
        {
            const int kr = tid >> 3, ns = (tid & 7) * 8;
            const float* src = B + (size_t)(k0 + kr) * N + n0 + ns;
            float4 f0 = ((const float4*)src)[0];
            float4 f1 = ((const float4*)src)[1];
            float tf[8] = {f0.x, f0.y, f0.z, f0.w, f1.x, f1.y, f1.z, f1.w};
            #pragma unroll
            for (int i = 0; i < 8; i++) Bl[ns + i][kr] = tob(tf[i]);
        }
        __syncthreads();

        v16bf bfrag = load_frag(&Bl[wn + lane16][8 * lhalf]);
        #pragma unroll
        for (int mt = 0; mt < 4; mt++) {
            v16bf afrag = load_frag(&Al[wm + mt * 16 + lane16][8 * lhalf]);
            acc[mt] = wmma_bf16(afrag, bfrag, acc[mt]);
        }
    }

    #pragma unroll
    for (int mt = 0; mt < 4; mt++) {
        #pragma unroll
        for (int r = 0; r < 8; r++) {
            int row = m0 + wm + mt * 16 + r + 8 * lhalf;
            C[(size_t)row * N + n0 + wn + lane16] = acc[mt][r];
        }
    }
}

// ---------------------------------------------------------------------------
// RoPE: q_raw (512 x 2048) -> Qw bf16 in (b,h,s,d) layout; k_raw -> Kn fp32.
// ---------------------------------------------------------------------------
__global__ __launch_bounds__(256) void rope_kernel(
    const float* __restrict__ q_raw, const float* __restrict__ k_raw,
    const float* __restrict__ cosp, const float* __restrict__ sinp,
    __bf16* __restrict__ Qw, float* __restrict__ Kn)
{
    int idx = blockIdx.x * 256 + threadIdx.x;      // 0 .. 512*2048-1
    int col = idx & 2047, token = idx >> 11;
    int h = col >> 8, d = col & 255;
    int b = token >> 5, s = token & 31;

    float c  = cosp[(size_t)token * 256 + d];
    float sn = sinp[(size_t)token * 256 + d];
    int dp = (d < 128) ? d + 128 : d - 128;

    float q  = q_raw[(size_t)token * 2048 + col];
    float qp = q_raw[(size_t)token * 2048 + h * 256 + dp];
    float rot = (d < 128) ? -qp : qp;
    Qw[(((size_t)b * 8 + h) * 32 + s) * 256 + d] = tob(q * c + rot * sn);

    if (h == 0) {
        float k  = k_raw[(size_t)token * 256 + d];
        float kp = k_raw[(size_t)token * 256 + dp];
        float kr = (d < 128) ? -kp : kp;
        Kn[(size_t)token * 256 + d] = k * c + kr * sn;
    }
}

// ---------------------------------------------------------------------------
// Split-KV flash attention. Grid = B*8 (b, kv-chunk). Block = 512 (16 waves).
// 129 kv-steps of 32 rows; 4096 = 128*32 so every step has a single source
// (cache for steps 0..127, new RoPE'd rows for step 128).
// Per step: TDM DMA of K (32x256 fp32, padded) into LDS; V staged transposed
// bf16; per wave 16 QK^T wmma + online softmax + 16 PV wmma.
// ---------------------------------------------------------------------------
__global__ __launch_bounds__(512) void attn_kernel(
    const float* __restrict__ Kc, const float* __restrict__ Vc,     // caches
    const __bf16* __restrict__ Qw,
    const float* __restrict__ Kn, const float* __restrict__ Vn,     // new k/v
    float* __restrict__ Opart, float* __restrict__ mpart, float* __restrict__ lpart)
{
    __shared__ float  Kf[32][260];          // TDM dest: 260-DWORD padded rows
    __shared__ __bf16 Vlds[256][40];        // transposed: [dim][kv], 80B stride
    __shared__ __bf16 Plds[16][16][40];     // per-wave P relayout scratch

    const int tid = threadIdx.x;
    const int w = tid >> 5, l = tid & 31;
    const int lane16 = l & 15, lhalf = l >> 4;
    const int b = blockIdx.x >> 3, chunk = blockIdx.x & 7;
    const int h = w >> 1, sr0 = (w & 1) << 4;

    // Q fragments: 16 rows x 256 dims, resident for the whole kernel
    v16bf qa[8];
    const __bf16* qbase = Qw + (((size_t)b * 8 + h) * 32 + sr0 + lane16) * 256;
    #pragma unroll
    for (int ks = 0; ks < 8; ks++) qa[ks] = load_frag(qbase + ks * 32 + 8 * lhalf);

    float m_r[8], l_r[8];
    #pragma unroll
    for (int r = 0; r < 8; r++) { m_r[r] = -1e30f; l_r[r] = 0.0f; }
    v8f acc[16] = {};

    const float qk_scale = 0.0625f * 1.44269504f;   // 1/sqrt(256) * log2(e)
    const int s0 = (129 * chunk) >> 3, s1 = (129 * (chunk + 1)) >> 3;

    for (int st = s0; st < s1; ++st) {
        const int kr0 = st * 32;
        __syncthreads();   // previous step's LDS reads complete

        // ---- K tile: single TDM descriptor issued by one wave
#if HAVE_TDM
        if (tid == 0) {
            const float* kbase = (kr0 < 4096)
                ? (Kc + ((size_t)b * 4096 + kr0) * 256)
                : (Kn + ((size_t)b * 32) * 256);
            tdm_load_k_tile(lds_addr_of(&Kf[0][0]), kbase, 32u);
            __builtin_amdgcn_s_wait_tensorcnt(0);
        }
#else
        {   // fallback: cooperative staging of K (fp32, padded rows)
            const int row = tid >> 4, seg = tid & 15, d0 = seg * 16;
            const int t = kr0 + row;
            const float* ksrc = (t < 4096)
                ? (Kc + ((size_t)b * 4096 + t) * 256 + d0)
                : (Kn + ((size_t)b * 32 + (t - 4096)) * 256 + d0);
            #pragma unroll
            for (int i = 0; i < 4; i++)
                ((float4*)&Kf[row][d0])[i] = ((const float4*)ksrc)[i];
        }
#endif
        // ---- V tile: convert fp32 -> bf16 and transpose into LDS
        {
            const int row = tid >> 4, seg = tid & 15, d0 = seg * 16;
            const int t = kr0 + row;
            const float* vsrc = (t < 4096)
                ? (Vc + ((size_t)b * 4096 + t) * 256 + d0)
                : (Vn + ((size_t)b * 32 + (t - 4096)) * 256 + d0);
            if (t + 32 < 4096)                           // stream-ahead hint
                __builtin_prefetch(vsrc + 32 * 256, 0, 1);
            float vt[16];
            #pragma unroll
            for (int i = 0; i < 4; i++) {
                float4 fv = ((const float4*)vsrc)[i];
                vt[4*i+0]=fv.x; vt[4*i+1]=fv.y; vt[4*i+2]=fv.z; vt[4*i+3]=fv.w;
            }
            #pragma unroll
            for (int i = 0; i < 16; i++) Vlds[d0 + i][row] = tob(vt[i]);
        }
        __syncthreads();

        // ---- scores: S(16x32) = Q(16x256) @ K^T(256x32)
        v8f sc0 = {}, sc1 = {};
        #pragma unroll
        for (int ks = 0; ks < 8; ks++) {
            v16bf kb0 = load_frag_f32(&Kf[lane16][ks * 32 + 8 * lhalf]);
            v16bf kb1 = load_frag_f32(&Kf[16 + lane16][ks * 32 + 8 * lhalf]);
            sc0 = wmma_bf16(qa[ks], kb0, sc0);
            sc1 = wmma_bf16(qa[ks], kb1, sc1);
        }

        // ---- online softmax (rows spread over 16-lane groups)
        float p0[8], p1[8], alpha[8];
        #pragma unroll
        for (int r = 0; r < 8; r++) {
            float v0 = sc0[r] * qk_scale, v1 = sc1[r] * qk_scale;
            float rm = fmaxf(v0, v1);
            #pragma unroll
            for (int mk = 1; mk < 16; mk <<= 1) rm = fmaxf(rm, __shfl_xor(rm, mk, 32));
            float mn = fmaxf(m_r[r], rm);
            alpha[r] = exp2f(m_r[r] - mn);
            m_r[r] = mn;
            p0[r] = exp2f(v0 - mn);
            p1[r] = exp2f(v1 - mn);
            float rs = p0[r] + p1[r];
            #pragma unroll
            for (int mk = 1; mk < 16; mk <<= 1) rs += __shfl_xor(rs, mk, 32);
            l_r[r] = l_r[r] * alpha[r] + rs;
        }
        #pragma unroll
        for (int nb = 0; nb < 16; nb++)
            #pragma unroll
            for (int r = 0; r < 8; r++) acc[nb][r] *= alpha[r];

        // ---- P: C layout -> A layout via wave-private LDS (in-order DS)
        #pragma unroll
        for (int r = 0; r < 8; r++) {
            int prow = r + 8 * lhalf;
            Plds[w][prow][lane16]      = tob(p0[r]);
            Plds[w][prow][16 + lane16] = tob(p1[r]);
        }
        v16bf pa = load_frag(&Plds[w][lane16][8 * lhalf]);

        // ---- O(16x256) += P(16x32) @ V(32x256)
        #pragma unroll
        for (int nb = 0; nb < 16; nb++) {
            v16bf vb = load_frag(&Vlds[nb * 16 + lane16][8 * lhalf]);
            acc[nb] = wmma_bf16(pa, vb, acc[nb]);
        }
    }

    // ---- write chunk partials
    const size_t basep = ((size_t)b * 8 + chunk) * 256;
    #pragma unroll
    for (int r = 0; r < 8; r++) {
        int qrow = h * 32 + sr0 + r + 8 * lhalf;
        #pragma unroll
        for (int nb = 0; nb < 16; nb++)
            Opart[(basep + qrow) * 256 + nb * 16 + lane16] = acc[nb][r];
        if (lane16 == 0) {
            mpart[basep + qrow] = m_r[r];
            lpart[basep + qrow] = l_r[r];
        }
    }
}

// ---------------------------------------------------------------------------
// Combine 8 chunk partials per (b,row); write attn output (B,S,H*D) fp32.
// ---------------------------------------------------------------------------
__global__ __launch_bounds__(256) void combine_kernel(
    const float* __restrict__ Opart, const float* __restrict__ mpart,
    const float* __restrict__ lpart, float* __restrict__ attnO)
{
    const int b = blockIdx.x >> 8, row = blockIdx.x & 255;
    float mc[8], m = -1e30f;
    #pragma unroll
    for (int c = 0; c < 8; c++) {
        mc[c] = mpart[((size_t)b * 8 + c) * 256 + row];
        m = fmaxf(m, mc[c]);
    }
    float wgt[8], L = 0.0f;
    #pragma unroll
    for (int c = 0; c < 8; c++) {
        wgt[c] = exp2f(mc[c] - m);
        L += lpart[((size_t)b * 8 + c) * 256 + row] * wgt[c];
    }
    const float inv = 1.0f / L;
    const int d = threadIdx.x;
    float o = 0.0f;
    #pragma unroll
    for (int c = 0; c < 8; c++)
        o += Opart[(((size_t)b * 8 + c) * 256 + row) * 256 + d] * wgt[c];
    const int h = row >> 5, s = row & 31;
    attnO[((size_t)b * 32 + s) * 2048 + h * 256 + d] = o * inv;
}

// ---------------------------------------------------------------------------
extern "C" void kernel_launch(void* const* d_in, const int* in_sizes, int n_in,
                              void* d_out, int out_size, void* d_ws, size_t ws_size,
                              hipStream_t stream) {
    (void)in_sizes; (void)n_in; (void)out_size; (void)ws_size;
    const float* hs   = (const float*)d_in[0];
    const float* cosp = (const float*)d_in[1];
    const float* sinp = (const float*)d_in[2];
    const float* Kc   = (const float*)d_in[3];
    const float* Vc   = (const float*)d_in[4];
    const float* Wq   = (const float*)d_in[5];
    const float* Wk   = (const float*)d_in[6];
    const float* Wv   = (const float*)d_in[7];
    const float* Wo   = (const float*)d_in[8];
    float* out = (float*)d_out;

    char* wsb = (char*)d_ws; size_t off = 0;
    auto alloc = [&](size_t bytes) -> void* {
        void* p = (void*)(wsb + off);
        off += (bytes + 255) & ~(size_t)255;
        return p;
    };
    float*  q_raw = (float*)alloc(512ull * 2048 * 4);
    float*  k_raw = (float*)alloc(512ull * 256 * 4);
    float*  v_raw = (float*)alloc(512ull * 256 * 4);
    float*  Kn    = (float*)alloc(512ull * 256 * 4);
    __bf16* Qw    = (__bf16*)alloc(512ull * 2048 * 2);
    float*  Opart = (float*)alloc(16ull * 8 * 256 * 256 * 4);
    float*  mpart = (float*)alloc(16ull * 8 * 256 * 4);
    float*  lpart = (float*)alloc(16ull * 8 * 256 * 4);
    float*  attnO = (float*)alloc(512ull * 2048 * 4);

    gemm_bf16_kernel<<<dim3(4, 32), 256, 0, stream>>>(hs, Wq, q_raw, 512, 2048, 1024);
    gemm_bf16_kernel<<<dim3(4, 4),  256, 0, stream>>>(hs, Wk, k_raw, 512, 256, 1024);
    gemm_bf16_kernel<<<dim3(4, 4),  256, 0, stream>>>(hs, Wv, v_raw, 512, 256, 1024);
    rope_kernel<<<4096, 256, 0, stream>>>(q_raw, k_raw, cosp, sinp, Qw, Kn);
    attn_kernel<<<128, 512, 0, stream>>>(Kc, Vc, Qw, Kn, v_raw, Opart, mpart, lpart);
    combine_kernel<<<4096, 256, 0, stream>>>(Opart, mpart, lpart, attnO);
    gemm_bf16_kernel<<<dim3(4, 16), 256, 0, stream>>>(attnO, Wo, out, 512, 1024, 2048);
}
